// AttentionBlock_17961553231883
// MI455X (gfx1250) — matrix operand order repaired and checked by reading, not compiled, not run on hardware
//
#include <hip/hip_runtime.h>
#include <math.h>

typedef __attribute__((ext_vector_type(16))) _Float16 v16h;
typedef __attribute__((ext_vector_type(2)))  _Float16 v2h;
typedef __attribute__((ext_vector_type(8)))  float    v8f;

// ---------------------------------------------------------------------------
// WMMA helpers. All operands are stored in fragment-major ("swizzled") form:
// a 16x32 A tile / 32x16 B tile occupies 512 halves; lane L's 16 fragment
// elements are the 32 contiguous bytes at tile_base + L*16 halves.
// ---------------------------------------------------------------------------
__device__ inline v8f zero8() {
    v8f z;
#pragma unroll
    for (int i = 0; i < 8; ++i) z[i] = 0.0f;
    return z;
}

__device__ inline v8f wmma_f16(v16h a, v16h b, v8f c) {
    return __builtin_amdgcn_wmma_f32_16x16x32_f16(false, a, false, b, (short)0, c, false, false);
}

// One 256-bit load per lane (lowers to 2x b128)
__device__ inline v16h load_frag(const _Float16* p) {
    return *(const v16h*)(p + (size_t)(threadIdx.x & 31) * 16);
}

// B fragment where only K rows 0..15 exist (stored compactly in lanes 0-15,
// 256 halves per tile); lanes 16-31 (K=16..31) are zero in registers.
__device__ inline v16h load_fragB_k16(const _Float16* p) {
    int lane = threadIdx.x & 31;
    v16h r;
    if (lane < 16) {
        r = *(const v16h*)(p + (size_t)lane * 16);
    } else {
#pragma unroll
        for (int i = 0; i < 16; ++i) r[i] = (_Float16)0.0f;
    }
    return r;
}

// Swizzled address of element (m,k) of an A-consumed matrix; KT = K/32 tiles.
__device__ inline size_t aswz(int KT, int m, int k) {
    size_t tile = (size_t)((m >> 4) * KT + (k >> 5));
    int half = (k >> 4) & 1;
    int e    = k & 15;
    return (tile << 9) + (size_t)(((half << 4) | (m & 15)) << 4) + e;
}

// Swizzled address of element (k,n) of a B-consumed matrix; NT = N/16 tiles.
__device__ inline size_t bswz(int NT, int k, int n) {
    size_t tile = (size_t)((k >> 5) * NT + (n >> 4));
    int lane = (((k >> 4) & 1) << 4) | (n & 15);
    return (tile << 9) + (size_t)(lane << 4) + (k & 15);
}

// Packed 2-half store (addresses proven 4B-aligned at call sites)
__device__ inline void store_h2(_Float16* p, _Float16 a, _Float16 b) {
    v2h t; t[0] = a; t[1] = b;
    *(v2h*)p = t;
}

// ---------------------------------------------------------------------------
// 16-lane-row butterfly reductions via v_permlane16_b32 (VALU only).
// ---------------------------------------------------------------------------
#if __has_builtin(__builtin_amdgcn_permlane16)
template <unsigned LO, unsigned HI>
__device__ inline float plane16(float x) {
    unsigned u = __float_as_uint(x);
    unsigned r = __builtin_amdgcn_permlane16(u, u, LO, HI, false, false);
    return __uint_as_float(r);
}
__device__ inline float half_max(float x) {
    x = fmaxf(x, plane16<0x67452301u, 0xEFCDAB89u>(x));   // xor 1
    x = fmaxf(x, plane16<0x54761032u, 0xDCFE98BAu>(x));   // xor 2
    x = fmaxf(x, plane16<0x32107654u, 0xBA98FEDCu>(x));   // xor 4
    x = fmaxf(x, plane16<0xFEDCBA98u, 0x76543210u>(x));   // xor 8
    return x;
}
__device__ inline float half_sum(float x) {
    x += plane16<0x67452301u, 0xEFCDAB89u>(x);
    x += plane16<0x54761032u, 0xDCFE98BAu>(x);
    x += plane16<0x32107654u, 0xBA98FEDCu>(x);
    x += plane16<0xFEDCBA98u, 0x76543210u>(x);
    return x;
}
#else
__device__ inline float half_max(float x) {
#pragma unroll
    for (int m = 1; m < 16; m <<= 1) x = fmaxf(x, __shfl_xor(x, m, 32));
    return x;
}
__device__ inline float half_sum(float x) {
#pragma unroll
    for (int m = 1; m < 16; m <<= 1) x += __shfl_xor(x, m, 32);
    return x;
}
#endif

// ---------------------------------------------------------------------------
// K0: f32 -> f16 weight conversion into A-swizzled layout (M x 192, KT=6)
// ---------------------------------------------------------------------------
__global__ void k0_convert_w(const float* __restrict__ src, _Float16* __restrict__ dst, int npairs) {
    int i = blockIdx.x * 256 + threadIdx.x;
    if (i < npairs) {
        int o = (2 * i) / 192, c = (2 * i) % 192;
        store_h2(dst + aswz(6, o, c), (_Float16)src[2 * i], (_Float16)src[2 * i + 1]);
    }
}

// ---------------------------------------------------------------------------
// K1: transpose x[b,c,t,hw] -> ybuf[c,32768] (residual) + GroupNorm(32)
// ---------------------------------------------------------------------------
__global__ void k1_norm_spatial(const float* __restrict__ x,
                                const float* __restrict__ gam,
                                const float* __restrict__ bet,
                                float* __restrict__ ybuf,
                                _Float16* __restrict__ xnh) {
    __shared__ float xb[6 * 1024];
    __shared__ float rs[256], rq[256];
    int n = blockIdx.x >> 5, grp = blockIdx.x & 31;
    int b = n >> 4, t = n & 15;
    int c0 = grp * 6;
    float s = 0.0f, s2 = 0.0f;
    for (int j = threadIdx.x; j < 6144; j += 256) {
        int cl = j >> 10, l = j & 1023;
        float v = x[((size_t)(b * 192 + c0 + cl) * 16 + t) * 1024 + l];
        xb[j] = v; s += v; s2 += v * v;
    }
    rs[threadIdx.x] = s; rq[threadIdx.x] = s2;
    __syncthreads();
    for (int st = 128; st > 0; st >>= 1) {
        if (threadIdx.x < st) { rs[threadIdx.x] += rs[threadIdx.x + st]; rq[threadIdx.x] += rq[threadIdx.x + st]; }
        __syncthreads();
    }
    float mu   = rs[0] * (1.0f / 6144.0f);
    float var  = rq[0] * (1.0f / 6144.0f) - mu * mu;
    float rinv = rsqrtf(var + 1e-5f);
    for (int j = threadIdx.x; j < 3072; j += 256) {
        int cp = j >> 10, l = j & 1023;       // channel pair 0..2
        int cl = cp * 2;
        int c  = c0 + cl;                     // even
        int col = n * 1024 + l;
        float v0 = xb[cl * 1024 + l];
        float v1 = xb[(cl + 1) * 1024 + l];
        ybuf[(size_t)c * 32768 + col]       = v0;
        ybuf[(size_t)(c + 1) * 32768 + col] = v1;
        store_h2(xnh + bswz(2048, c, col),
                 (_Float16)((v0 - mu) * rinv * gam[c] + bet[c]),
                 (_Float16)((v1 - mu) * rinv * gam[c + 1] + bet[c + 1]));
    }
}

// ---------------------------------------------------------------------------
// Shared WMMA GEMM: Y[M,32768] = Wh[M,192] @ X[192,32768]; all frag loads b128
// MODE 0: qkv spatial scatter   MODE 1: proj spatial + residual (in-place ybuf)
// MODE 2: qkv temporal scatter  MODE 3: proj temporal + residual + transpose out
// ---------------------------------------------------------------------------
template <int MODE>
__global__ void gemm_conv(const _Float16* __restrict__ Wh,
                          const _Float16* __restrict__ X,
                          const float* __restrict__ bias,
                          float* __restrict__ ybuf,
                          float* __restrict__ outf,
                          _Float16* __restrict__ oq,
                          _Float16* __restrict__ ok,
                          _Float16* __restrict__ ov) {
    int wave  = threadIdx.x >> 5;
    int strip = blockIdx.x * 8 + wave;
    int mtile = strip >> 9;                 // strip / 512
    int ct0   = (strip & 511) * 4;          // first of 4 column tiles
    int nl = threadIdx.x & 15, half = (threadIdx.x >> 4) & 1;

    v8f acc[4];
#pragma unroll
    for (int j = 0; j < 4; ++j) acc[j] = zero8();

#pragma unroll
    for (int kt = 0; kt < 6; ++kt) {
        v16h a = load_frag(Wh + ((size_t)(mtile * 6 + kt) << 9));
#pragma unroll
        for (int j = 0; j < 4; ++j) {
            v16h b = load_frag(X + ((size_t)(kt * 2048 + ct0 + j) << 9));
            acc[j] = wmma_f16(a, b, acc[j]);
        }
    }

    const float scale = 0.35355339059327373f;   // 64^-0.25
#pragma unroll
    for (int j = 0; j < 4; ++j) {
        int col = (ct0 + j) * 16 + nl;
#pragma unroll
        for (int v = 0; v < 8; v += 2) {            // o pairs: same part, cc even
            int o0 = mtile * 16 + v + 8 * half;
            float val0 = acc[j][v]     + bias[o0];
            float val1 = acc[j][v + 1] + bias[o0 + 1];
            if (MODE == 0 || MODE == 2) {
                int head = o0 / 192, rem = o0 % 192, part = rem / 64, cc = rem % 64;
                if (MODE == 0) {
                    int nb = col >> 10, l = col & 1023;
                    size_t base = (size_t)(nb * 3 + head) * 65536;
                    if (part == 0)
                        store_h2(oq + base + aswz(2, l, cc), (_Float16)(val0 * scale), (_Float16)(val1 * scale));
                    else if (part == 1)
                        store_h2(ok + base + bswz(64, cc, l), (_Float16)(val0 * scale), (_Float16)(val1 * scale));
                    else {
                        ov[base + bswz(4, l, cc)]     = (_Float16)val0;
                        ov[base + bswz(4, l, cc + 1)] = (_Float16)val1;
                    }
                } else {
                    int n2 = col >> 4, t = col & 15;
                    size_t base = (size_t)(n2 * 3 + head) * 1024;
                    if (part == 0)
                        store_h2(oq + base + aswz(2, t, cc), (_Float16)(val0 * scale), (_Float16)(val1 * scale));
                    else if (part == 1)
                        store_h2(ok + base + bswz(1, cc, t), (_Float16)(val0 * scale), (_Float16)(val1 * scale));
                    else {
                        ov[base + (size_t)(cc >> 4) * 256 + (size_t)(cc & 15) * 16 + t]             = (_Float16)val0;
                        ov[base + (size_t)((cc + 1) >> 4) * 256 + (size_t)((cc + 1) & 15) * 16 + t] = (_Float16)val1;
                    }
                }
            } else if (MODE == 1) {
                size_t i0 = (size_t)o0 * 32768 + col;
                ybuf[i0]         = val0 + ybuf[i0];
                ybuf[i0 + 32768] = val1 + ybuf[i0 + 32768];
            } else { // MODE 3
                int n2 = col >> 4, t = col & 15;
                int b = n2 >> 10, l = n2 & 1023;
                size_t rbase = (size_t)(b * 16 + t) * 1024 + l;
                outf[((size_t)(b * 192 + o0) * 16 + t) * 1024 + l]     = val0 + ybuf[(size_t)o0 * 32768 + rbase];
                outf[((size_t)(b * 192 + o0 + 1) * 16 + t) * 1024 + l] = val1 + ybuf[(size_t)(o0 + 1) * 32768 + rbase];
            }
        }
    }
}

// ---------------------------------------------------------------------------
// K3: spatial flash attention. One wave = 16 queries, online softmax over
//     64-column chunks. launch_bounds(128,1): keep Q frags + pointers in
//     registers (no spills); prefetch next chunk's K/V tiles.
// ---------------------------------------------------------------------------
__global__ void __launch_bounds__(128, 1)
k3_attn_spatial(const _Float16* __restrict__ qh,
                const _Float16* __restrict__ kh,
                const _Float16* __restrict__ vh,
                _Float16* __restrict__ ah) {
    __shared__ __align__(32) _Float16 pb[4 * 1024];   // 2 A tiles per wave
    int wave = threadIdx.x >> 5;
    int batch = blockIdx.x;
    int tblock = blockIdx.y * 4 + wave;
    int nl = threadIdx.x & 15, half = (threadIdx.x >> 4) & 1;
    int lane = threadIdx.x & 31;

    const _Float16* qb = qh + (size_t)batch * 65536;
    const _Float16* kb = kh + (size_t)batch * 65536;
    const _Float16* vb = vh + (size_t)batch * 65536;

    v16h aq0 = load_frag(qb + ((size_t)(tblock * 2 + 0) << 9));
    v16h aq1 = load_frag(qb + ((size_t)(tblock * 2 + 1) << 9));

    float M[8], L[8];
#pragma unroll
    for (int v = 0; v < 8; ++v) { M[v] = -3.0e38f; L[v] = 0.0f; }
    v8f O[4];
#pragma unroll
    for (int j = 0; j < 4; ++j) O[j] = zero8();

    _Float16* pw = pb + wave * 1024;

    for (int cs = 0; cs < 1024; cs += 64) {
        // prefetch first lines of next chunk's K/V tiles (wraps on last iter)
        int ncs = (cs + 64) & 1023;
        __builtin_prefetch(kb + ((size_t)(ncs >> 4) << 9) + (size_t)lane * 16, 0, 3);
        __builtin_prefetch(vb + ((size_t)((ncs >> 5) * 4) << 9) + (size_t)lane * 16, 0, 3);

        v8f S[4];
#pragma unroll
        for (int st = 0; st < 4; ++st) {
            int nt = (cs >> 4) + st;
            v8f a = zero8();
            a = wmma_f16(aq0, load_frag(kb + ((size_t)(0 * 64 + nt) << 9)), a);
            a = wmma_f16(aq1, load_frag(kb + ((size_t)(1 * 64 + nt) << 9)), a);
            S[st] = a;
        }
#pragma unroll
        for (int v = 0; v < 8; ++v) {
            float mx = fmaxf(fmaxf(S[0][v], S[1][v]), fmaxf(S[2][v], S[3][v]));
            mx = half_max(mx);
            float Mn = fmaxf(M[v], mx);
            float al = __expf(M[v] - Mn);
            float ss = 0.0f;
#pragma unroll
            for (int st = 0; st < 4; ++st) { float p = __expf(S[st][v] - Mn); S[st][v] = p; ss += p; }
            ss = half_sum(ss);
            L[v] = L[v] * al + ss;
            M[v] = Mn;
            O[0][v] *= al; O[1][v] *= al; O[2][v] *= al; O[3][v] *= al;
        }
        // D-layout -> A-swizzled LDS tiles (m = query row, k = local column)
#pragma unroll
        for (int st = 0; st < 4; ++st) {
#pragma unroll
            for (int v = 0; v < 8; ++v)
                pw[aswz(2, v + 8 * half, st * 16 + nl)] = (_Float16)S[st][v];
        }
        v16h ap0 = load_frag(pw);
        v16h ap1 = load_frag(pw + 512);
#pragma unroll
        for (int ct = 0; ct < 4; ++ct) {
            O[ct] = wmma_f16(ap0, load_frag(vb + ((size_t)(((cs >> 5) + 0) * 4 + ct) << 9)), O[ct]);
            O[ct] = wmma_f16(ap1, load_frag(vb + ((size_t)(((cs >> 5) + 1) * 4 + ct) << 9)), O[ct]);
        }
    }

    float rl[8];
#pragma unroll
    for (int v = 0; v < 8; ++v) rl[v] = __builtin_amdgcn_rcpf(L[v]);

    int nb = batch / 3, head = batch % 3;
#pragma unroll
    for (int ct = 0; ct < 4; ++ct) {
#pragma unroll
        for (int v = 0; v < 8; ++v) {
            int t = tblock * 16 + v + 8 * half;
            int row = head * 64 + ct * 16 + nl;
            ah[bswz(2048, row, nb * 1024 + t)] = (_Float16)(O[ct][v] * rl[v]);
        }
    }
}

// ---------------------------------------------------------------------------
// K5: temporal GroupNorm per (n2 = b*1024+hw, group) over 6 ch x 16 t
// ---------------------------------------------------------------------------
__global__ void k5_norm_temporal(const float* __restrict__ ybuf,
                                 const float* __restrict__ gam,
                                 const float* __restrict__ bet,
                                 _Float16* __restrict__ xnh) {
    int tid = blockIdx.x * 256 + threadIdx.x;
    if (tid >= 65536) return;
    int n2 = tid >> 5, grp = tid & 31;
    int b = n2 >> 10, l = n2 & 1023;
    int c0 = grp * 6;
    float s = 0.0f, s2 = 0.0f;
    for (int cl = 0; cl < 6; ++cl)
        for (int t = 0; t < 16; ++t) {
            float v = ybuf[(size_t)(c0 + cl) * 32768 + (size_t)(b * 16 + t) * 1024 + l];
            s += v; s2 += v * v;
        }
    float mu = s * (1.0f / 96.0f);
    float var = s2 * (1.0f / 96.0f) - mu * mu;
    float rinv = rsqrtf(var + 1e-5f);
    for (int cl = 0; cl < 6; cl += 2) {
        int c = c0 + cl;                       // even
        float g0 = gam[c], b0 = bet[c], g1 = gam[c + 1], b1 = bet[c + 1];
        for (int t = 0; t < 16; ++t) {
            float v0 = ybuf[(size_t)c * 32768 + (size_t)(b * 16 + t) * 1024 + l];
            float v1 = ybuf[(size_t)(c + 1) * 32768 + (size_t)(b * 16 + t) * 1024 + l];
            store_h2(xnh + bswz(2048, c, n2 * 16 + t),
                     (_Float16)((v0 - mu) * rinv * g0 + b0),
                     (_Float16)((v1 - mu) * rinv * g1 + b1));
        }
    }
}

// ---------------------------------------------------------------------------
// K7: temporal attention (L=16). One wave per (n2, head); K padded to 32.
// ---------------------------------------------------------------------------
__global__ void k7_attn_temporal(const _Float16* __restrict__ q2,
                                 const _Float16* __restrict__ k2,
                                 const _Float16* __restrict__ v2,
                                 _Float16* __restrict__ ah) {
    __shared__ __align__(32) _Float16 pb[8 * 512];
    int wave = threadIdx.x >> 5;
    int b2 = blockIdx.x * 8 + wave;
    int nl = threadIdx.x & 15, half = (threadIdx.x >> 4) & 1;
    int lane = threadIdx.x & 31;

    const _Float16* qb = q2 + (size_t)b2 * 1024;
    const _Float16* kb = k2 + (size_t)b2 * 1024;
    const _Float16* vb = v2 + (size_t)b2 * 1024;

    v16h aq0 = load_frag(qb);
    v16h aq1 = load_frag(qb + 512);
    v8f S = zero8();
    S = wmma_f16(aq0, load_frag(kb), S);
    S = wmma_f16(aq1, load_frag(kb + 512), S);

#pragma unroll
    for (int v = 0; v < 8; ++v) {
        float mx = half_max(S[v]);
        float p = __expf(S[v] - mx);
        float ss = half_sum(p);
        S[v] = p * __builtin_amdgcn_rcpf(ss);
    }

    _Float16* pw = pb + wave * 512;      // one A tile 16x32, k>=16 zero
#pragma unroll
    for (int i = 0; i < 4; ++i)          // lanes 16..31 (k>=16) zero, packed
        store_h2(pw + 256 + lane * 8 + 2 * i, (_Float16)0.0f, (_Float16)0.0f);
#pragma unroll
    for (int v = 0; v < 8; ++v) pw[aswz(1, v + 8 * half, nl)] = (_Float16)S[v];

    v16h ap = load_frag(pw);
    int n2 = b2 / 3, head = b2 % 3;
#pragma unroll
    for (int ct = 0; ct < 4; ++ct) {
        v8f O = wmma_f16(ap, load_fragB_k16(vb + ct * 256), zero8());
#pragma unroll
        for (int v = 0; v < 8; ++v) {
            int t = v + 8 * half;
            int row = head * 64 + ct * 16 + nl;
            ah[bswz(2048, row, n2 * 16 + t)] = (_Float16)O[v];
        }
    }
}

// ---------------------------------------------------------------------------
extern "C" void kernel_launch(void* const* d_in, const int* in_sizes, int n_in,
                              void* d_out, int out_size, void* d_ws, size_t ws_size,
                              hipStream_t stream) {
    const float* x       = (const float*)d_in[0];
    const float* norm_g  = (const float*)d_in[1];
    const float* norm_b  = (const float*)d_in[2];
    const float* qkv_w   = (const float*)d_in[3];
    const float* qkv_b   = (const float*)d_in[4];
    const float* proj_w  = (const float*)d_in[5];
    const float* proj_b  = (const float*)d_in[6];
    const float* normt_g = (const float*)d_in[7];
    const float* normt_b = (const float*)d_in[8];
    const float* qkvt_w  = (const float*)d_in[9];
    const float* qkvt_b  = (const float*)d_in[10];
    const float* projt_w = (const float*)d_in[11];
    const float* projt_b = (const float*)d_in[12];
    float* out = (float*)d_out;

    char* ws = (char*)d_ws;
    size_t off = 0;
    auto alloc = [&](size_t bytes) -> char* {
        char* p = ws + off;
        off = (off + bytes + 255) & ~(size_t)255;
        return p;
    };
    float*    ybuf   = (float*)   alloc(192UL * 32768 * 4);   // residual / y (in-place)
    _Float16* xnh    = (_Float16*)alloc(192UL * 32768 * 2);   // normalized, B-swz (reused)
    _Float16* ah     = (_Float16*)alloc(192UL * 32768 * 2);   // attention out, B-swz (reused)
    _Float16* qh     = (_Float16*)alloc(96UL * 65536 * 2);    // A-swz (reused as q2)
    _Float16* kh     = (_Float16*)alloc(96UL * 65536 * 2);    // B-swz (reused as k2)
    _Float16* vh     = (_Float16*)alloc(96UL * 65536 * 2);    // B-swz (reused as v2)
    _Float16* wqkvh  = (_Float16*)alloc(576UL * 192 * 2);
    _Float16* wprojh = (_Float16*)alloc(192UL * 192 * 2);
    _Float16* wqkvth = (_Float16*)alloc(576UL * 192 * 2);
    _Float16* wprojth= (_Float16*)alloc(192UL * 192 * 2);

    k0_convert_w<<<(55296 + 255) / 256, 256, 0, stream>>>(qkv_w,   wqkvh,   55296);
    k0_convert_w<<<(18432 + 255) / 256, 256, 0, stream>>>(proj_w,  wprojh,  18432);
    k0_convert_w<<<(55296 + 255) / 256, 256, 0, stream>>>(qkvt_w,  wqkvth,  55296);
    k0_convert_w<<<(18432 + 255) / 256, 256, 0, stream>>>(projt_w, wprojth, 18432);

    // ---- spatial pass ----
    k1_norm_spatial<<<1024, 256, 0, stream>>>(x, norm_g, norm_b, ybuf, xnh);
    gemm_conv<0><<<2304, 256, 0, stream>>>(wqkvh, xnh, qkv_b, nullptr, nullptr, qh, kh, vh);
    k3_attn_spatial<<<dim3(96, 16), 128, 0, stream>>>(qh, kh, vh, ah);
    gemm_conv<1><<<768, 256, 0, stream>>>(wprojh, ah, proj_b, ybuf, nullptr, nullptr, nullptr, nullptr);

    // ---- temporal pass ----
    k5_norm_temporal<<<256, 256, 0, stream>>>(ybuf, normt_g, normt_b, xnh);
    gemm_conv<2><<<2304, 256, 0, stream>>>(wqkvth, xnh, qkvt_b, nullptr, nullptr, qh, kh, vh);
    k7_attn_temporal<<<768, 256, 0, stream>>>(qh, kh, vh, ah);
    gemm_conv<3><<<768, 256, 0, stream>>>(wprojth, ah, projt_b, ybuf, out, nullptr, nullptr, nullptr);
}
